// PerResidueEncoder_42949672961888
// MI455X (gfx1250) — compile-verified, matrix operand order checked by reading, not run on hardware
//
#include <hip/hip_runtime.h>
#include <hip/hip_bf16.h>
#include <math.h>

// ---------------------------------------------------------------------------
// Types
// ---------------------------------------------------------------------------
typedef __bf16 bf16_t;
typedef __attribute__((ext_vector_type(16))) __bf16 v16bf;
typedef __attribute__((ext_vector_type(8)))  __bf16 v8bf;
typedef __attribute__((ext_vector_type(8)))  float  v8f;

// ---------------------------------------------------------------------------
// Problem constants (from reference)
// ---------------------------------------------------------------------------
static constexpr int Bsz      = 32;
static constexpr int Lsz      = 512;
static constexpr int NTOK     = Bsz * Lsz;        // 16384 tokens
static constexpr int FEAT     = 128;
static constexpr int ESM_DIM  = 1280;
static constexpr int INFEAT   = 726;              // 128 + 78 + 520
static constexpr int INFEAT_P = 736;              // padded to multiple of 32
static constexpr int HID      = 256;              // FEAT*2

// ---------------------------------------------------------------------------
// Weight packing: f32 W[K][Nout] (fan_in x fan_out, row-major) -> bf16
// fragments matching the CDNA5 16-bit operand layout (lane<16 holds col n,
// K offsets {0..7,16..23}; lane>=16 holds K offsets {8..15,24..31}).
// Zero-pads K rows beyond the real K.
// Fragment address: ((nt*ktiles + kt)*32 + lane)*16 elements.
// ---------------------------------------------------------------------------
__global__ void pack_w_kernel(const float* __restrict__ W, bf16_t* __restrict__ out,
                              int K, int Kpad, int Nout) {
    const int ktiles = Kpad >> 5;
    const int ntiles = Nout >> 4;
    const int total  = ntiles * ktiles * 32;
    int idx = blockIdx.x * blockDim.x + threadIdx.x;
    if (idx >= total) return;
    const int lane = idx & 31;
    const int kt   = (idx >> 5) % ktiles;
    const int nt   = idx / (32 * ktiles);
    const int n    = nt * 16 + (lane & 15);
    const int kbase = kt * 32 + ((lane >> 4) ? 8 : 0);
    bf16_t* o = out + (size_t)idx * 16;
#pragma unroll
    for (int e = 0; e < 16; ++e) {
        const int k = kbase + (e < 8 ? e : e + 8);   // chunk2 lives at +16
        float v = (k < K) ? W[(size_t)k * Nout + n] : 0.0f;
        o[e] = (bf16_t)v;
    }
}

// ---------------------------------------------------------------------------
// Fragment loaders
// ---------------------------------------------------------------------------
template<bool AF32>
__device__ __forceinline__ v16bf load_a_frag(const void* __restrict__ Aany,
                                             size_t rowA, int K, int kt, int half) {
    v16bf a;
    if (AF32) {
        const float* ap = (const float*)Aany + rowA * (size_t)K + kt * 32 + half * 8;
        v8f f0 = *(const v8f*)(ap);
        v8f f1 = *(const v8f*)(ap + 16);
#pragma unroll
        for (int e = 0; e < 8; ++e) { a[e] = (bf16_t)f0[e]; a[8 + e] = (bf16_t)f1[e]; }
    } else {
        const bf16_t* ap = (const bf16_t*)Aany + rowA * (size_t)K + kt * 32 + half * 8;
        v8bf a0 = *(const v8bf*)(ap);
        v8bf a1 = *(const v8bf*)(ap + 16);
#pragma unroll
        for (int e = 0; e < 8; ++e) { a[e] = a0[e]; a[8 + e] = a1[e]; }
    }
    return a;
}

// ---------------------------------------------------------------------------
// Fused bf16 WMMA GEMM: C[M x Nout] = act(A[M x K] * Bpacked + bias) * rowscale
//   - 256 threads = 8 waves; wave w owns rows [blockIdx.y*128 + w*16, +16)
//   - each wave register-blocks 4 N-tiles (64 cols) -> 4 WMMAs per K-step
//   - software-pipelined: loads for kt+1 issue before the WMMAs of kt, so
//     s_wait only covers one iteration of in-flight loads
//   - AF32: read A as f32 and convert inline (first ESM layer)
//   - GELU_ACT: exact gelu (erf) epilogue
// ---------------------------------------------------------------------------
template<bool AF32, bool GELU_ACT>
__global__ __launch_bounds__(256)
void gemm_wmma_kernel(const void*  __restrict__ Aany,
                      const bf16_t* __restrict__ Bp,
                      const float*  __restrict__ bias,
                      const float*  __restrict__ rowscale,
                      bf16_t* __restrict__ out_bf,
                      float*  __restrict__ out_f32,
                      int K, int ldc) {
    const int lane   = threadIdx.x & 31;
    const int wave   = threadIdx.x >> 5;
    const int ktiles = K >> 5;
    const int m_base = (blockIdx.y * 8 + wave) * 16;
    const int n_base = blockIdx.x * 64;
    const int half   = lane >> 4;        // which K-half this lane holds
    const int l15    = lane & 15;

    v8f acc[4];
#pragma unroll
    for (int t = 0; t < 4; ++t) { v8f z = {}; acc[t] = z; }

    const size_t bstride = (size_t)ktiles * 512;          // elems per n-tile
    const bf16_t* bcol = Bp + (size_t)(n_base >> 4) * bstride + (size_t)lane * 16;
    const size_t rowA = (size_t)(m_base + l15);

    // ---- prologue: load fragments for kt = 0 ----
    v16bf a_cur = load_a_frag<AF32>(Aany, rowA, K, 0, half);
    v16bf b_cur[4];
#pragma unroll
    for (int t = 0; t < 4; ++t) b_cur[t] = *(const v16bf*)(bcol + (size_t)t * bstride);

    // ---- pipelined main loop: fetch kt+1 while computing kt ----
#pragma unroll 2
    for (int kt = 0; kt < ktiles - 1; ++kt) {
        const bf16_t* bpn = bcol + (size_t)(kt + 1) * 512;
        v16bf a_nxt = load_a_frag<AF32>(Aany, rowA, K, kt + 1, half);
        v16bf b_nxt[4];
#pragma unroll
        for (int t = 0; t < 4; ++t) b_nxt[t] = *(const v16bf*)(bpn + (size_t)t * bstride);
        __builtin_prefetch(bpn + 512, 0, 3);               // kt+2, tile 0 (near scope)
        __builtin_prefetch(bpn + 512 + 2 * bstride, 0, 3); // kt+2, tile 2
#pragma unroll
        for (int t = 0; t < 4; ++t)
            acc[t] = __builtin_amdgcn_wmma_f32_16x16x32_bf16(
                         false, a_cur, false, b_cur[t], (short)0, acc[t], false, false);
        a_cur = a_nxt;
#pragma unroll
        for (int t = 0; t < 4; ++t) b_cur[t] = b_nxt[t];
    }
    // ---- epilogue WMMAs for the last K-step ----
#pragma unroll
    for (int t = 0; t < 4; ++t)
        acc[t] = __builtin_amdgcn_wmma_f32_16x16x32_bf16(
                     false, a_cur, false, b_cur[t], (short)0, acc[t], false, false);

    // ---- epilogue: bias + (gelu) + rowscale, store bf16 / f32 ----
#pragma unroll
    for (int t = 0; t < 4; ++t) {
        const int col = n_base + t * 16 + l15;
        const float bv = bias[col];
#pragma unroll
        for (int r = 0; r < 8; ++r) {
            const int row = m_base + r + half * 8;
            float v = acc[t][r] + bv;
            if (GELU_ACT) v = 0.5f * v * (1.0f + erff(v * 0.70710678118654752f));
            if (rowscale) v *= rowscale[row];
            if (out_bf)  out_bf [(size_t)row * ldc + col] = (bf16_t)v;
            if (out_f32) out_f32[(size_t)row * ldc + col] = v;
        }
    }
}

// ---------------------------------------------------------------------------
// Geometry features: RBF(65 pairs x 8) + dihedral encodings (6 x 13) + zero pad
// One block per token (128 threads). Writes bf16 row of width 736; cols 0..127
// (aa_feat) are written in-place by the ESM layer-2 GEMM.
// ---------------------------------------------------------------------------
__device__ __forceinline__ void pair_ij(int p, int& i, int& j) {
    const int PI[10] = {1, 1, 1, 1, 0, 0, 0, 4, 4, 3};
    const int PJ[10] = {0, 2, 3, 4, 2, 3, 4, 2, 3, 2};
    if (p < 10) { i = PI[p]; j = PJ[p]; return; }
    int q = p - 10, cnt = 10;
    i = 4;
    while (q >= cnt) { q -= cnt; ++i; --cnt; }
    j = i + 1 + q;
}

__global__ void feat_kernel(const float* __restrict__ X,          // [N,15,3]
                            const float* __restrict__ mask_atom,  // [N,15]
                            const float* __restrict__ phi, const float* __restrict__ phi_mask,
                            const float* __restrict__ psi, const float* __restrict__ psi_mask,
                            const float* __restrict__ chi, const float* __restrict__ chi_mask,
                            const float* __restrict__ mask_res,
                            bf16_t* __restrict__ feat) {          // [N,736]
    const int t   = blockIdx.x;
    const int tid = threadIdx.x;
    const float mres = mask_res[t];
    bf16_t* frow = feat + (size_t)t * INFEAT_P;

    if (tid < 65) {
        int i, j; pair_ij(tid, i, j);
        const float* Xt = X + (size_t)t * 45;
        const float dx = Xt[i * 3 + 0] - Xt[j * 3 + 0];
        const float dy = Xt[i * 3 + 1] - Xt[j * 3 + 1];
        const float dz = Xt[i * 3 + 2] - Xt[j * 3 + 2];
        const float D = sqrtf(dx * dx + dy * dy + dz * dz + 1e-6f);
        float pm = 1.0f;
        if (tid >= 10) pm = mask_atom[(size_t)t * 15 + i] * mask_atom[(size_t)t * 15 + j];
        pm *= mres;
#pragma unroll
        for (int r = 0; r < 8; ++r) {
            const float mu = 0.2f + (float)r * (6.0f / 7.0f);
            const float z  = (D - mu) * (1.0f / 0.75f);
            frow[206 + tid * 8 + r] = (bf16_t)(expf(-z * z) * pm);
        }
    } else if (tid < 71) {
        const int d = tid - 65;
        float x, dm;
        if      (d == 0) { x = phi[t]; dm = phi_mask[t]; }
        else if (d == 1) { x = psi[t]; dm = psi_mask[t]; }
        else             { x = chi[(size_t)t * 4 + (d - 2)]; dm = chi_mask[(size_t)t * 4 + (d - 2)]; }
        const float s = dm * mres;
        const float F[6] = {1.0f, 2.0f, 3.0f, 1.0f, 0.5f, 1.0f / 3.0f};
        bf16_t* o = frow + 128 + d * 13;
        o[0] = (bf16_t)(x * s);
#pragma unroll
        for (int f = 0; f < 6; ++f) {
            o[1 + f] = (bf16_t)(sinf(x * F[f]) * s);
            o[7 + f] = (bf16_t)(cosf(x * F[f]) * s);
        }
    } else if (tid < 81) {
        frow[INFEAT + (tid - 71)] = (bf16_t)0.0f;   // zero pad cols 726..735
    }
}

// ---------------------------------------------------------------------------
// Workspace layout (bytes, 256-aligned)
// ---------------------------------------------------------------------------
static constexpr size_t OFF_H     = 0;                                     // [N,1280] bf16
static constexpr size_t OFF_FEAT  = OFF_H    + (size_t)NTOK * ESM_DIM * 2; // [N,736] bf16
static constexpr size_t OFF_Y1    = OFF_FEAT + (size_t)NTOK * INFEAT_P * 2;
static constexpr size_t OFF_Y2    = OFF_Y1   + (size_t)NTOK * HID * 2;
static constexpr size_t OFF_Y3    = OFF_Y2   + (size_t)NTOK * FEAT * 2;
static constexpr size_t OFF_EW1   = OFF_Y3   + (size_t)NTOK * FEAT * 2;
static constexpr size_t OFF_EW2   = OFF_EW1  + (size_t)ESM_DIM * ESM_DIM * 2;
static constexpr size_t OFF_MW1   = OFF_EW2  + (size_t)ESM_DIM * FEAT * 2;
static constexpr size_t OFF_MW2   = OFF_MW1  + (size_t)INFEAT_P * HID * 2;
static constexpr size_t OFF_MW3   = OFF_MW2  + (size_t)HID * FEAT * 2;
static constexpr size_t OFF_MW4   = OFF_MW3  + (size_t)FEAT * FEAT * 2;

static inline int pack_blocks(int Kpad, int Nout) {
    int total = (Nout / 16) * (Kpad / 32) * 32;
    return (total + 255) / 256;
}

extern "C" void kernel_launch(void* const* d_in, const int* in_sizes, int n_in,
                              void* d_out, int out_size, void* d_ws, size_t ws_size,
                              hipStream_t stream) {
    (void)in_sizes; (void)n_in; (void)out_size; (void)ws_size;

    const float* aa_esm2   = (const float*)d_in[1];
    const float* X         = (const float*)d_in[2];
    const float* mask_atom = (const float*)d_in[3];
    const float* phi       = (const float*)d_in[4];
    const float* phi_mask  = (const float*)d_in[5];
    const float* psi       = (const float*)d_in[6];
    const float* psi_mask  = (const float*)d_in[7];
    const float* chi       = (const float*)d_in[8];
    const float* chi_mask  = (const float*)d_in[9];
    const float* mask_res  = (const float*)d_in[10];
    const float* esm_W1 = (const float*)d_in[11]; const float* esm_b1 = (const float*)d_in[12];
    const float* esm_W2 = (const float*)d_in[13]; const float* esm_b2 = (const float*)d_in[14];
    const float* mlp_W1 = (const float*)d_in[15]; const float* mlp_b1 = (const float*)d_in[16];
    const float* mlp_W2 = (const float*)d_in[17]; const float* mlp_b2 = (const float*)d_in[18];
    const float* mlp_W3 = (const float*)d_in[19]; const float* mlp_b3 = (const float*)d_in[20];
    const float* mlp_W4 = (const float*)d_in[21]; const float* mlp_b4 = (const float*)d_in[22];

    char* ws = (char*)d_ws;
    bf16_t* Hbf    = (bf16_t*)(ws + OFF_H);
    bf16_t* featbf = (bf16_t*)(ws + OFF_FEAT);
    bf16_t* y1bf   = (bf16_t*)(ws + OFF_Y1);
    bf16_t* y2bf   = (bf16_t*)(ws + OFF_Y2);
    bf16_t* y3bf   = (bf16_t*)(ws + OFF_Y3);
    bf16_t* eW1p   = (bf16_t*)(ws + OFF_EW1);
    bf16_t* eW2p   = (bf16_t*)(ws + OFF_EW2);
    bf16_t* mW1p   = (bf16_t*)(ws + OFF_MW1);
    bf16_t* mW2p   = (bf16_t*)(ws + OFF_MW2);
    bf16_t* mW3p   = (bf16_t*)(ws + OFF_MW3);
    bf16_t* mW4p   = (bf16_t*)(ws + OFF_MW4);
    float*  out    = (float*)d_out;

    // --- pack all weights to WMMA fragment layout (bf16, K zero-padded) ---
    pack_w_kernel<<<pack_blocks(ESM_DIM, ESM_DIM), 256, 0, stream>>>(esm_W1, eW1p, ESM_DIM, ESM_DIM, ESM_DIM);
    pack_w_kernel<<<pack_blocks(ESM_DIM, FEAT),    256, 0, stream>>>(esm_W2, eW2p, ESM_DIM, ESM_DIM, FEAT);
    pack_w_kernel<<<pack_blocks(INFEAT_P, HID),    256, 0, stream>>>(mlp_W1, mW1p, INFEAT, INFEAT_P, HID);
    pack_w_kernel<<<pack_blocks(HID, FEAT),        256, 0, stream>>>(mlp_W2, mW2p, HID, HID, FEAT);
    pack_w_kernel<<<pack_blocks(FEAT, FEAT),       256, 0, stream>>>(mlp_W3, mW3p, FEAT, FEAT, FEAT);
    pack_w_kernel<<<pack_blocks(FEAT, FEAT),       256, 0, stream>>>(mlp_W4, mW4p, FEAT, FEAT, FEAT);

    // --- geometry / dihedral features into feat[:, 128:736] ---
    feat_kernel<<<NTOK, 128, 0, stream>>>(X, mask_atom, phi, phi_mask, psi, psi_mask,
                                          chi, chi_mask, mask_res, featbf);

    const int MB = NTOK / 128;   // 128 row-blocks (8 waves x 16 rows each)

    // --- ESM layer 1: gelu(esm2 @ W1 + b1) -> H (bf16) ---
    gemm_wmma_kernel<true, true><<<dim3(ESM_DIM / 64, MB), 256, 0, stream>>>(
        aa_esm2, eW1p, esm_b1, nullptr, Hbf, nullptr, ESM_DIM, ESM_DIM);

    // --- ESM layer 2: (H @ W2 + b2) * mask -> feat[:, 0:128] (ldc = 736) ---
    gemm_wmma_kernel<false, false><<<dim3(FEAT / 64, MB), 256, 0, stream>>>(
        Hbf, eW2p, esm_b2, mask_res, featbf, nullptr, ESM_DIM, INFEAT_P);

    // --- MLP: 736 -> 256 -> 128 -> 128 -> 128 ---
    gemm_wmma_kernel<false, true><<<dim3(HID / 64, MB), 256, 0, stream>>>(
        featbf, mW1p, mlp_b1, nullptr, y1bf, nullptr, INFEAT_P, HID);
    gemm_wmma_kernel<false, true><<<dim3(FEAT / 64, MB), 256, 0, stream>>>(
        y1bf, mW2p, mlp_b2, nullptr, y2bf, nullptr, HID, FEAT);
    gemm_wmma_kernel<false, true><<<dim3(FEAT / 64, MB), 256, 0, stream>>>(
        y2bf, mW3p, mlp_b3, nullptr, y3bf, nullptr, FEAT, FEAT);
    gemm_wmma_kernel<false, false><<<dim3(FEAT / 64, MB), 256, 0, stream>>>(
        y3bf, mW4p, mlp_b4, mask_res, nullptr, out, FEAT, FEAT);
}
